// sgcn_14250701488881
// MI455X (gfx1250) — matrix operand order repaired and checked by reading, not compiled, not run on hardware
//
#include <hip/hip_runtime.h>
#include <hip/hip_bf16.h>
#include <math.h>

#define NNODES 100000
#define NEDGES 3200000
#define INDIM  256
#define LDH    160   // padded feature stride (5*32) for intermediate buffers

typedef __bf16 bf16_t;
typedef bf16_t v16bf __attribute__((ext_vector_type(16)));
typedef float  v8f   __attribute__((ext_vector_type(8)));

// ---------------------------------------------------------------------------
// zero a small float array (per-call determinism: stats must start at 0)
// ---------------------------------------------------------------------------
__global__ void zero_f32_kernel(float* p, int n) {
    int i = blockIdx.x * blockDim.x + threadIdx.x;
    if (i < n) p[i] = 0.f;
}

// ---------------------------------------------------------------------------
// column sums / sums-of-squares of x [N,256] for batchnorm folding
// ---------------------------------------------------------------------------
__global__ void colstats_kernel(const float* __restrict__ x,
                                float* __restrict__ sum, float* __restrict__ sumsq,
                                int nrows) {
    int c  = threadIdx.x;            // 256 threads = 256 columns
    int r0 = blockIdx.x * 256;
    int rend = r0 + 256; if (rend > nrows) rend = nrows;
    float s = 0.f, s2 = 0.f;
    for (int r = r0; r < rend; ++r) {
        float v = x[(size_t)r * INDIM + c];
        s += v; s2 += v * v;
    }
    atomicAdd(&sum[c], s);
    atomicAdd(&sumsq[c], s2);
}

// ---------------------------------------------------------------------------
// fold batchnorm into W_in/b_in:  (x*sc + sh) @ W = x @ (sc*W) + sh @ W
// ---------------------------------------------------------------------------
__global__ void fold_bn_kernel(const float* __restrict__ sum, const float* __restrict__ sumsq,
                               const float* __restrict__ g, const float* __restrict__ b,
                               const float* __restrict__ Win, const float* __restrict__ bin,
                               float* __restrict__ Wf, float* __restrict__ bf) {
    int j = threadIdx.x;
    if (j >= 140) return;
    const float invN = 1.f / (float)NNODES;
    float acc = bin[j];
    for (int c = 0; c < INDIM; ++c) {
        float mu  = sum[c]   * invN;
        float var = sumsq[c] * invN - mu * mu;
        float sc  = g[c] * rsqrtf(var + 1e-5f);
        float sh  = b[c] - mu * sc;
        float w   = Win[c * 140 + j];
        Wf[c * 140 + j] = sc * w;
        acc += sh * w;
    }
    bf[j] = acc;
}

// ---------------------------------------------------------------------------
// pack weight W[K][Nout] (row-major, ld=Nout) into transposed, zero-padded
// bf16 Wt[Npad][Kpad] so each lane's WMMA B-fragment is one contiguous load.
// ---------------------------------------------------------------------------
__global__ void pack_wt_kernel(const float* __restrict__ W, int K, int Nout,
                               bf16_t* __restrict__ Wt, int Kpad, int Npad) {
    int idx = blockIdx.x * blockDim.x + threadIdx.x;
    if (idx >= Kpad * Npad) return;
    int n = idx / Kpad;
    int k = idx - n * Kpad;
    float v = (k < K && n < Nout) ? W[(size_t)k * Nout + n] : 0.f;
    Wt[idx] = (bf16_t)v;
}

// ---------------------------------------------------------------------------
// CSR row pointers from sorted rows[] via lower_bound binary search
// ---------------------------------------------------------------------------
__global__ void build_rowptr_kernel(const int* __restrict__ rows, int* __restrict__ rowptr,
                                    int nrows, int nedges) {
    int r = blockIdx.x * blockDim.x + threadIdx.x;
    if (r > nrows) return;
    int lo = 0, hi = nedges;
    while (lo < hi) {
        int mid = (lo + hi) >> 1;
        if (rows[mid] < r) lo = mid + 1; else hi = mid;
    }
    rowptr[r] = lo;
}

// ---------------------------------------------------------------------------
// WMMA GEMM: C[M,Nout] = A[M,Kpad] @ Wt^T + bias, optional tanh.
// A is f32 with zero-padded columns up to Kpad (multiple of 32), so the
// K loop is branch-free: 4x float4 loads (A) + 1x 32B load (B bf16) + wmma.
// Columns [Nout, Nfill) of C are zero-filled to maintain the padding
// invariant for downstream consumers. C/D VGPR layout per CDNA5 ISA 7.12.2; A/B use contiguous k = 16*half + s.
// ---------------------------------------------------------------------------
__global__ void gemm_wmma_kernel(const float* __restrict__ A, int lda,
                                 const bf16_t* __restrict__ Wt, int Kpad,
                                 const float* __restrict__ bias,
                                 float* __restrict__ C, int ldc,
                                 int Nout, int Nfill, int act) {
    const int lane = threadIdx.x & 31;
    const int wave = threadIdx.x >> 5;
    const int m0   = blockIdx.x * 16;
    const int n0   = (blockIdx.y * 4 + wave) * 16;
    if (n0 >= Nfill) return;                 // wave-uniform

    const int half = lane >> 4;              // 0: lanes 0-15, 1: lanes 16-31
    const int l15  = lane & 15;
    const int bN   = n0 + l15;

    if (n0 >= Nout) {                        // pure zero-fill wave (wave-uniform)
        if (bN < Nfill) {
            #pragma unroll
            for (int v = 0; v < 8; ++v)
                C[(size_t)(m0 + v + half * 8) * ldc + bN] = 0.f;
        }
        return;
    }

    const int aK = half * 16;                // A K-offset per half-wave
    const int bK = half * 16;                // B K-offset per half-wave

    const float*  __restrict__ Arow = A  + (size_t)(m0 + l15) * lda;
    const bf16_t* __restrict__ Wrow = Wt + (size_t)bN * Kpad;   // bN < Npad guaranteed

    v8f acc = {0.f, 0.f, 0.f, 0.f, 0.f, 0.f, 0.f, 0.f};

    #pragma unroll 2
    for (int k = 0; k < Kpad; k += 32) {
        // A fragment: 16 contiguous-in-pairs floats -> bf16
        float af[16];
        *(float4*)(af + 0)  = *(const float4*)(Arow + k + aK);
        *(float4*)(af + 4)  = *(const float4*)(Arow + k + aK + 4);
        *(float4*)(af + 8)  = *(const float4*)(Arow + k + aK + 8);
        *(float4*)(af + 12) = *(const float4*)(Arow + k + aK + 12);
        v16bf a;
        #pragma unroll
        for (int j = 0; j < 16; ++j) a[j] = (bf16_t)af[j];

        // B fragment: 16 contiguous bf16 (32B, aligned)
        v16bf bm = *(const v16bf*)(Wrow + k + bK);

        acc = __builtin_amdgcn_wmma_f32_16x16x32_bf16(
            /*neg_a=*/false, a, /*neg_b=*/false, bm,
            /*c_mod=*/(short)0, acc, /*reuse_a=*/false, /*reuse_b=*/false);
    }

    if (bN < Nout) {
        float bv = bias[bN];
        #pragma unroll
        for (int v = 0; v < 8; ++v) {        // C/D layout: VGPR v -> M = v + 8*half
            int row = m0 + v + half * 8;
            float y = acc[v] + bv;
            if (act == 1) y = tanhf(y);
            C[(size_t)row * ldc + bN] = y;
        }
    } else if (bN < Nfill) {
        #pragma unroll
        for (int v = 0; v < 8; ++v)
            C[(size_t)(m0 + v + half * 8) * ldc + bN] = 0.f;
    }
}

// ---------------------------------------------------------------------------
// SpMM hop: out[r,:] = sum_{e in row r} vals[e] * in[cols[e],:]
// One wave32 per destination row; edges broadcast via shfl (uniform lane ->
// readlane + SGPR-operand FMA). Guard-free: operates on NSEG*32 columns,
// relying on the zero-padding invariant of the buffers.
// ---------------------------------------------------------------------------
template <int NSEG>
__global__ void spmm_kernel(const int* __restrict__ rowptr,
                            const int* __restrict__ cols,
                            const float* __restrict__ vals,
                            const float* __restrict__ xin,
                            float* __restrict__ xout,
                            int nrows) {
    const int lane = threadIdx.x & 31;
    const int wave = threadIdx.x >> 5;
    const int row  = blockIdx.x * 8 + wave;
    if (row >= nrows) return;

    const int s = rowptr[row];
    const int e = rowptr[row + 1];

    float acc[NSEG];
    #pragma unroll
    for (int j = 0; j < NSEG; ++j) acc[j] = 0.f;

    for (int base = s; base < e; base += 32) {
        if (base + 32 < e) {                          // prefetch next edge chunk
            __builtin_prefetch(&cols[base + 32 + lane], 0, 1);
            __builtin_prefetch(&vals[base + 32 + lane], 0, 1);
        }
        int nE = e - base; if (nE > 32) nE = 32;
        int   c = 0; float v = 0.f;
        if (lane < nE) { c = cols[base + lane]; v = vals[base + lane]; }
        for (int i = 0; i < nE; ++i) {
            int   ci = __shfl(c, i, 32);
            float vi = __shfl(v, i, 32);
            const float* __restrict__ xr = xin + (size_t)ci * LDH;
            #pragma unroll
            for (int j = 0; j < NSEG; ++j)
                acc[j] += vi * xr[lane + 32 * j];
        }
    }
    float* __restrict__ orow = xout + (size_t)row * LDH;
    #pragma unroll
    for (int j = 0; j < NSEG; ++j) orow[lane + 32 * j] = acc[j];
}

// ---------------------------------------------------------------------------
// LayerNorm + tanh, in place, one wave32 per row (F <= 120 -> 4 regs/lane)
// ---------------------------------------------------------------------------
__global__ void ln_tanh_kernel(float* __restrict__ h,
                               const float* __restrict__ g, const float* __restrict__ b,
                               int F, int nrows) {
    const int lane = threadIdx.x & 31;
    const int wave = threadIdx.x >> 5;
    const int row  = blockIdx.x * 8 + wave;
    if (row >= nrows) return;

    float* __restrict__ hr = h + (size_t)row * LDH;
    float v[4];
    float s = 0.f;
    #pragma unroll
    for (int j = 0; j < 4; ++j) {
        int f = lane + 32 * j;
        v[j] = (f < F) ? hr[f] : 0.f;
        s += v[j];
    }
    for (int o = 16; o > 0; o >>= 1) s += __shfl_xor(s, o, 32);
    float mu = s / (float)F;

    float q = 0.f;
    #pragma unroll
    for (int j = 0; j < 4; ++j) {
        int f = lane + 32 * j;
        float d = (f < F) ? (v[j] - mu) : 0.f;
        q += d * d;
    }
    for (int o = 16; o > 0; o >>= 1) q += __shfl_xor(q, o, 32);
    float rs = rsqrtf(q / (float)F + 1e-5f);

    #pragma unroll
    for (int j = 0; j < 4; ++j) {
        int f = lane + 32 * j;
        if (f < F) hr[f] = tanhf((v[j] - mu) * rs * g[f] + b[f]);
    }
}

// ---------------------------------------------------------------------------
extern "C" void kernel_launch(void* const* d_in, const int* in_sizes, int n_in,
                              void* d_out, int out_size, void* d_ws, size_t ws_size,
                              hipStream_t stream) {
    (void)in_sizes; (void)n_in; (void)out_size; (void)ws_size;

    const float* x        = (const float*)d_in[0];
    const float* adj_vals = (const float*)d_in[1];
    const float* bn_g     = (const float*)d_in[2];
    const float* bn_b     = (const float*)d_in[3];
    const float* W_in     = (const float*)d_in[4];
    const float* b_in     = (const float*)d_in[5];
    const float* W1       = (const float*)d_in[6];
    const float* b1       = (const float*)d_in[7];
    const float* ln1_g    = (const float*)d_in[8];
    const float* ln1_b    = (const float*)d_in[9];
    const float* W2       = (const float*)d_in[10];
    const float* b2       = (const float*)d_in[11];
    const float* ln2_g    = (const float*)d_in[12];
    const float* ln2_b    = (const float*)d_in[13];
    const float* W_out    = (const float*)d_in[14];
    const float* b_out    = (const float*)d_in[15];
    const int*   adj_rows = (const int*)d_in[16];
    const int*   adj_cols = (const int*)d_in[17];
    float*       out      = (float*)d_out;

    // workspace layout
    auto align_up = [](size_t v) { return (v + 255) & ~(size_t)255; };
    size_t off = 0;
    char* ws = (char*)d_ws;
    float*  d_sum   = (float*) (ws + off); off = align_up(off + 256 * sizeof(float));
    float*  d_sumsq = (float*) (ws + off); off = align_up(off + 256 * sizeof(float));
    float*  d_Wf    = (float*) (ws + off); off = align_up(off + (size_t)256 * 140 * sizeof(float));
    float*  d_bf    = (float*) (ws + off); off = align_up(off + 140 * sizeof(float));
    int*    d_rp    = (int*)   (ws + off); off = align_up(off + (size_t)(NNODES + 1) * sizeof(int));
    bf16_t* d_WtIn  = (bf16_t*)(ws + off); off = align_up(off + (size_t)144 * 256 * sizeof(bf16_t));
    bf16_t* d_Wt1   = (bf16_t*)(ws + off); off = align_up(off + (size_t)128 * 160 * sizeof(bf16_t));
    bf16_t* d_Wt2   = (bf16_t*)(ws + off); off = align_up(off + (size_t)112 * 128 * sizeof(bf16_t));
    bf16_t* d_WtOut = (bf16_t*)(ws + off); off = align_up(off + (size_t)64  * 128 * sizeof(bf16_t));
    float*  bufA    = (float*) (ws + off); off = align_up(off + (size_t)NNODES * LDH * sizeof(float));
    float*  bufB    = (float*) (ws + off); off = align_up(off + (size_t)NNODES * LDH * sizeof(float));

    const int MT  = NNODES / 16;             // 6250 exact M tiles
    const int SPB = (NNODES + 7) / 8;        // 12500 blocks for wave-per-row kernels

    // 1) batchnorm stats + fold into first linear layer (d_sum/d_sumsq contiguous)
    zero_f32_kernel<<<1, 512, 0, stream>>>(d_sum, 512);
    colstats_kernel<<<(NNODES + 255) / 256, 256, 0, stream>>>(x, d_sum, d_sumsq, NNODES);
    fold_bn_kernel<<<1, 160, 0, stream>>>(d_sum, d_sumsq, bn_g, bn_b, W_in, b_in, d_Wf, d_bf);

    // 2) pack weights: transposed, zero-padded bf16
    pack_wt_kernel<<<(144 * 256 + 255) / 256, 256, 0, stream>>>(d_Wf,  256, 140, d_WtIn,  256, 144);
    pack_wt_kernel<<<(128 * 160 + 255) / 256, 256, 0, stream>>>(W1,    140, 120, d_Wt1,   160, 128);
    pack_wt_kernel<<<(112 * 128 + 255) / 256, 256, 0, stream>>>(W2,    120, 100, d_Wt2,   128, 112);
    pack_wt_kernel<<<(64  * 128 + 255) / 256, 256, 0, stream>>>(W_out, 100, 64,  d_WtOut, 128, 64);

    // 3) CSR row pointers
    build_rowptr_kernel<<<(NNODES + 1 + 255) / 256, 256, 0, stream>>>(adj_rows, d_rp, NNODES, NEDGES);

    // 4) h = tanh(x @ Wf + bf)  [N,140], zero-fill cols 140..159
    gemm_wmma_kernel<<<dim3(MT, 3), 128, 0, stream>>>(x, INDIM, d_WtIn, 256, d_bf,
                                                      bufA, LDH, 140, 160, /*tanh*/1);

    // 5) 4 SpMM hops at F=140 (160-wide, zeros propagate)  A->B->A->B->A
    spmm_kernel<5><<<SPB, 256, 0, stream>>>(d_rp, adj_cols, adj_vals, bufA, bufB, NNODES);
    spmm_kernel<5><<<SPB, 256, 0, stream>>>(d_rp, adj_cols, adj_vals, bufB, bufA, NNODES);
    spmm_kernel<5><<<SPB, 256, 0, stream>>>(d_rp, adj_cols, adj_vals, bufA, bufB, NNODES);
    spmm_kernel<5><<<SPB, 256, 0, stream>>>(d_rp, adj_cols, adj_vals, bufB, bufA, NNODES);

    // 6) h = h @ W1 + b1 (K=140->Kpad=160) ; zero-fill to 160 ; LN + tanh  [N,120]
    gemm_wmma_kernel<<<dim3(MT, 3), 128, 0, stream>>>(bufA, LDH, d_Wt1, 160, b1,
                                                      bufB, LDH, 120, 160, 0);
    ln_tanh_kernel<<<SPB, 256, 0, stream>>>(bufB, ln1_g, ln1_b, 120, NNODES);

    // 7) 4 SpMM hops at F=120 (128-wide)  B->A->B->A->B
    spmm_kernel<4><<<SPB, 256, 0, stream>>>(d_rp, adj_cols, adj_vals, bufB, bufA, NNODES);
    spmm_kernel<4><<<SPB, 256, 0, stream>>>(d_rp, adj_cols, adj_vals, bufA, bufB, NNODES);
    spmm_kernel<4><<<SPB, 256, 0, stream>>>(d_rp, adj_cols, adj_vals, bufB, bufA, NNODES);
    spmm_kernel<4><<<SPB, 256, 0, stream>>>(d_rp, adj_cols, adj_vals, bufA, bufB, NNODES);

    // 8) h = h @ W2 + b2 (K=120->Kpad=128) ; zero-fill to 128 ; LN + tanh  [N,100]
    gemm_wmma_kernel<<<dim3(MT, 2), 128, 0, stream>>>(bufB, LDH, d_Wt2, 128, b2,
                                                      bufA, LDH, 100, 128, 0);
    ln_tanh_kernel<<<SPB, 256, 0, stream>>>(bufA, ln2_g, ln2_b, 100, NNODES);

    // 9) out = h @ W_out + b_out (K=100->Kpad=128)  [N,64]
    gemm_wmma_kernel<<<dim3(MT, 1), 128, 0, stream>>>(bufA, LDH, d_WtOut, 128, b_out,
                                                      out, 64, 64, 64, 0);
}
